// InstanceSegmentationLoss_15049565405344
// MI455X (gfx1250) — compile-verified, matrix operand order checked by reading, not run on hardware
//
#include <hip/hip_runtime.h>

// MI455X / gfx1250, wave32. Confusion-matrix-as-GEMM with V_WMMA_I32_16X16X64_IU8
// on one-hot int8 operands, then a tiny IoU/loss reduction kernel.

typedef __attribute__((ext_vector_type(8))) int v8i;

#define IDS 64
#define HISTN (IDS * IDS)
#define MAIN_BLOCKS 128
#define MAIN_THREADS 256   // 8 wave32 per workgroup

// ---------------------------------------------------------------------------
// Kernel 0: zero the 64x64 i32 histogram in workspace.
// ---------------------------------------------------------------------------
__global__ void zero_hist_kernel(int* __restrict__ hist) {
    int i = blockIdx.x * blockDim.x + threadIdx.x;
    if (i < HISTN) hist[i] = 0;
}

// ---------------------------------------------------------------------------
// Exact per-byte equality -> 0x01 one-hot bytes.
// packed bytes and idrep bytes are both < 0x40, so x+0x7f7f7f7f has no
// cross-byte carries; bit7 of each byte of t is set iff that byte of x != 0.
// ---------------------------------------------------------------------------
__device__ __forceinline__ unsigned int onehot_bytes(unsigned int packed,
                                                     unsigned int idrep) {
    unsigned int x = packed ^ idrep;
    unsigned int t = x + 0x7f7f7f7fu;
    return ((~t) >> 7) & 0x01010101u;   // 0x01 where byte matched
}

// ---------------------------------------------------------------------------
// Kernel 1: joint histogram of (pred_id, true_id) via IU8 WMMA.
// Each wave owns the full 64x64 id space (16 accumulator blocks) over its
// pixel stripe; per 64-pixel chunk it builds one-hot A (pred ids) and B
// (true ids) operands and issues 16 v_wmma_i32_16x16x64_iu8.
// ---------------------------------------------------------------------------
__global__ __launch_bounds__(MAIN_THREADS)
void confusion_wmma_kernel(const int* __restrict__ pred,
                           const int* __restrict__ truem,
                           int* __restrict__ hist, int n) {
    // Per-wave staging slot: qwords 0..7 = 64 pred id bytes, 8..15 = true bytes.
    __shared__ unsigned long long sbuf[MAIN_THREADS / 32][16];

    const int lane = threadIdx.x & 31;
    const int wv   = threadIdx.x >> 5;
    const int gw   = blockIdx.x * (MAIN_THREADS / 32) + wv;
    const int totalWaves = gridDim.x * (MAIN_THREADS / 32);
    const int half = lane >> 4;   // 0: A/B K-bytes {16v..16v+7}, 1: {16v+8..16v+15}
    const int sub  = lane & 15;   // row id (A) / col id (B) within a 16-block

    v8i acc[4][4];
#pragma unroll
    for (int mb = 0; mb < 4; ++mb)
#pragma unroll
        for (int nb = 0; nb < 4; ++nb)
#pragma unroll
            for (int r = 0; r < 8; ++r) acc[mb][nb][r] = 0;

    unsigned short* sp = (unsigned short*)&sbuf[wv][0];

    for (int base = gw * 64; base + 64 <= n; base += totalWaves * 64) {
        // 64 pixels per wave: 2 consecutive ids per lane (coalesced b64 loads).
        int2 p2 = *(const int2*)(pred  + base + 2 * lane);
        int2 t2 = *(const int2*)(truem + base + 2 * lane);
        sp[lane]      = (unsigned short)((p2.x & 63) | ((p2.y & 63) << 8));
        sp[32 + lane] = (unsigned short)((t2.x & 63) | ((t2.y & 63) << 8));
        // LDS is in-order within a wave; fence compiler + drain DS stores.
        asm volatile("s_wait_dscnt 0" ::: "memory");

        // Gather the qwords this lane contributes to the IU8 operand layout.
        unsigned long long aq[4], bq[4];
#pragma unroll
        for (int v = 0; v < 4; ++v) {
            aq[v] = sbuf[wv][2 * v + half];
            bq[v] = sbuf[wv][8 + 2 * v + half];
        }

        // Build all 4 B one-hot blocks once (reused by every mb).
        v8i bmat[4];
#pragma unroll
        for (int b = 0; b < 4; ++b) {
            unsigned int rep = (unsigned int)(b * 16 + sub) * 0x01010101u;
#pragma unroll
            for (int v = 0; v < 4; ++v) {
                bmat[b][2 * v]     = (int)onehot_bytes((unsigned int)bq[v], rep);
                bmat[b][2 * v + 1] = (int)onehot_bytes((unsigned int)(bq[v] >> 32), rep);
            }
        }

#pragma unroll
        for (int mb = 0; mb < 4; ++mb) {
            unsigned int rep = (unsigned int)(mb * 16 + sub) * 0x01010101u;
            v8i amat;
#pragma unroll
            for (int v = 0; v < 4; ++v) {
                amat[2 * v]     = (int)onehot_bytes((unsigned int)aq[v], rep);
                amat[2 * v + 1] = (int)onehot_bytes((unsigned int)(aq[v] >> 32), rep);
            }
#pragma unroll
            for (int nb = 0; nb < 4; ++nb) {
                acc[mb][nb] = __builtin_amdgcn_wmma_i32_16x16x64_iu8(
                    /*sgn_a=*/false, amat, /*sgn_b=*/false, bmat[nb],
                    acc[mb][nb], /*reuse_a=*/false, /*reuse_b=*/false);
            }
        }
    }

    // Flush: C/D layout -> VGPR r holds M=r (lanes 0-15) / M=r+8 (lanes 16-31),
    // N = lane&15. Skip zero entries (blocks are sparse for uniform ids).
#pragma unroll
    for (int mb = 0; mb < 4; ++mb)
#pragma unroll
        for (int nb = 0; nb < 4; ++nb)
#pragma unroll
            for (int r = 0; r < 8; ++r) {
                int v = acc[mb][nb][r];
                if (v != 0) {
                    int m  = mb * 16 + half * 8 + r;
                    int nn = nb * 16 + sub;
                    atomicAdd(&hist[m * IDS + nn], v);
                }
            }
}

// ---------------------------------------------------------------------------
// Kernel 2: 63x63 IoU + row/col maxes + final scalar. One block, 64 threads.
// area_p/area_t are full row/col sums of the 64x64 histogram (every pixel has
// some true/pred id, including background), matching pm.sum / tm.sum.
// ---------------------------------------------------------------------------
__global__ void loss_kernel(const int* __restrict__ hist, float* __restrict__ out) {
    __shared__ float areaP[IDS], areaT[IDS], red[IDS];
    const int t = threadIdx.x;  // 0..63

    int rs = 0, cs = 0;
    for (int j = 0; j < IDS; ++j) {
        rs += hist[t * IDS + j];
        cs += hist[j * IDS + t];
    }
    areaP[t] = (float)rs;
    areaT[t] = (float)cs;
    __syncthreads();

    float rowmax = 0.0f, colmax = 0.0f;
    if (t >= 1) {
        for (int j = 1; j < IDS; ++j) {
            float ir = (float)hist[t * IDS + j];
            float un = areaP[t] + areaT[j] - ir;
            float iou = (un == 0.0f) ? 0.0f : ir / un;
            rowmax = fmaxf(rowmax, iou);

            float ic = (float)hist[j * IDS + t];
            float un2 = areaP[j] + areaT[t] - ic;
            float iou2 = (un2 == 0.0f) ? 0.0f : ic / un2;
            colmax = fmaxf(colmax, iou2);
        }
    }
    red[t] = (t >= 1) ? (2.0f - rowmax - colmax) : 0.0f;
    __syncthreads();
    for (int s = 32; s > 0; s >>= 1) {
        if (t < s) red[t] += red[t + s];
        __syncthreads();
    }
    if (t == 0) out[0] = red[0] / (float)(2 * (IDS - 1));
}

// ---------------------------------------------------------------------------
extern "C" void kernel_launch(void* const* d_in, const int* in_sizes, int n_in,
                              void* d_out, int out_size, void* d_ws, size_t ws_size,
                              hipStream_t stream) {
    const int* pred  = (const int*)d_in[0];
    const int* truem = (const int*)d_in[1];
    int* hist = (int*)d_ws;           // 4096 i32 = 16 KB scratch
    float* out = (float*)d_out;
    const int n = in_sizes[0];        // H*W = 1048576

    zero_hist_kernel<<<(HISTN + 255) / 256, 256, 0, stream>>>(hist);
    confusion_wmma_kernel<<<MAIN_BLOCKS, MAIN_THREADS, 0, stream>>>(pred, truem, hist, n);
    loss_kernel<<<1, IDS, 0, stream>>>(hist, out);
}
